// LATEFUSION_IMG_TRANS_VSLT_GRUD_65420941853096
// MI455X (gfx1250) — compile-verified, matrix operand order checked by reading, not run on hardware
//
#include <hip/hip_runtime.h>

// ---------------------------------------------------------------------------
// Problem constants (match reference)
// ---------------------------------------------------------------------------
#define B_   4096
#define T_   48
#define F_   16
#define D_   256
#define G_   768          // 3*D
#define K_   12
#define H_   64
#define CPAD 544          // C=514 padded to 17*32
#define BN_EPS 1e-5f

typedef __attribute__((ext_vector_type(16))) __bf16 v16bf;
typedef __attribute__((ext_vector_type(8)))  float  v8f;
typedef __attribute__((ext_vector_type(4)))  unsigned int v4u;
typedef __attribute__((ext_vector_type(8)))  int v8i;
typedef __attribute__((ext_vector_type(4)))  int v4i;

struct __attribute__((aligned(16))) U4 { unsigned int x, y, z, w; };
union FragB { v16bf v; U4 u[2]; };

__device__ inline v8f wmma_bf16(v16bf a, v16bf b, v8f c) {
    // D = A(16x32 bf16) * B(32x16 bf16) + C(16x16 f32)
    return __builtin_amdgcn_wmma_f32_16x16x32_bf16(
        false, a, false, b, (short)0, c, false, false);
}

__device__ inline v8f splat8(float x) {
    v8f v;
#pragma unroll
    for (int i = 0; i < 8; ++i) v[i] = x;
    return v;
}

__device__ inline float sigmf(float x) { return 1.0f / (1.0f + __expf(-x)); }

// ---------------------------------------------------------------------------
// TDM: async 2-D tile load (global -> LDS), bf16 elements.
//   D# per CDNA5 ISA 8.3/8.4: group0 = {count=1 | lds_addr | global_addr |
//   type=2}; group1 = {data_size=2B, tensor dims, tile dims, dim0 stride}.
//   tileW/rowStride in elements; tileW*2 must be a multiple of 4 bytes.
//   This toolchain exposes the 6-arg builtin: (g0, g1, g2, g3, g_extra, cpol).
// ---------------------------------------------------------------------------
__device__ inline void tdm_load_2d_bf16(unsigned ldsOff, const void* gptr,
                                        unsigned tileW, unsigned tileH,
                                        unsigned rowStride)
{
    unsigned long long ga = (unsigned long long)gptr;
    v4u g0;
    g0[0] = 1u;                                        // count=1, user descriptor
    g0[1] = ldsOff;                                    // lds_addr (bytes)
    g0[2] = (unsigned)(ga & 0xFFFFFFFFu);              // global_addr[31:0]
    g0[3] = (unsigned)((ga >> 32) & 0x1FFFFFFu)        // global_addr[56:32]
          | (2u << 30);                                // type = 2 ("image")
    v8i g1;
    g1[0] = (int)(1u << 16);                           // data_size=1 (2 bytes), wg_mask=0
    g1[1] = (int)((rowStride & 0xFFFFu) << 16);        // tensor_dim0[15:0]
    g1[2] = (int)(((rowStride >> 16) & 0xFFFFu)        // tensor_dim0[31:16]
          | ((tileH & 0xFFFFu) << 16));                // tensor_dim1[15:0]
    g1[3] = (int)((tileW & 0xFFFFu) << 16);            // tensor_dim1[31:16]=0 | tile_dim0
    g1[4] = (int)(tileH & 0xFFFFu);                    // tile_dim1 (tile_dim2=0)
    g1[5] = (int)rowStride;                            // tensor_dim0_stride[31:0]
    g1[6] = 0;                                         // stride hi / dim1_stride
    g1[7] = 0;
    v4i z4 = {0, 0, 0, 0};
    v8i z8 = {0, 0, 0, 0, 0, 0, 0, 0};
    __builtin_amdgcn_tensor_load_to_lds(g0, g1, z4, z4, z8, 0);
}

// ---------------------------------------------------------------------------
// Kernel 1: elementwise GRU-D input preprocessing -> bf16 [xx|m] and d
// ---------------------------------------------------------------------------
__global__ __launch_bounds__(256) void prep_x_kernel(
    const float* __restrict__ x, const float* __restrict__ m,
    const float* __restrict__ d, const float* __restrict__ x_m,
    const float* __restrict__ dec_w, const float* __restrict__ dec_b,
    __bf16* __restrict__ Xc, __bf16* __restrict__ Db)
{
    size_t i = (size_t)blockIdx.x * 256 + threadIdx.x;
    if (i >= (size_t)B_ * T_ * F_) return;
    int f = (int)(i & (F_ - 1));
    size_t bt = i >> 4;
    float dv = d[i], mv = m[i], xv = x[i];
    float xd = __expf(-fmaxf(dv * dec_w[f] + dec_b[f], 0.0f));
    float xx = mv * xv + (1.0f - mv) * xd * xv + (1.0f - mv) * (1.0f - xd) * x_m[f];
    Xc[bt * 32 + f]      = (__bf16)xx;
    Xc[bt * 32 + 16 + f] = (__bf16)mv;
    Db[bt * 16 + f]      = (__bf16)dv;
}

// ---------------------------------------------------------------------------
// Kernel 2: weight conversion f32 -> bf16 ([N][K] layouts; fc_W1 zero-padded)
// ---------------------------------------------------------------------------
__global__ __launch_bounds__(256) void prep_w_kernel(
    const float* __restrict__ W_ih, const float* __restrict__ W_hh,
    const float* __restrict__ hd_W, const float* __restrict__ fc_W1,
    __bf16* __restrict__ Wih, __bf16* __restrict__ Whh,
    __bf16* __restrict__ hdW, __bf16* __restrict__ W1b)
{
    int i = blockIdx.x * 256 + threadIdx.x;
    const int n1 = G_ * 32, n2 = G_ * D_, n3 = D_ * F_, n4 = K_ * H_ * CPAD;
    if (i < n1) { Wih[i] = (__bf16)W_ih[i]; return; }
    i -= n1;
    if (i < n2) { Whh[i] = (__bf16)W_hh[i]; return; }
    i -= n2;
    if (i < n3) { hdW[i] = (__bf16)hd_W[i]; return; }
    i -= n3;
    if (i < n4) {
        int c = i % CPAD, kh = i / CPAD;   // kh in [0, K_*H_)
        W1b[i] = (c < 514) ? (__bf16)fc_W1[(size_t)kh * 514 + c] : (__bf16)0.0f;
    }
}

// ---------------------------------------------------------------------------
// Kernel 3: persistent-per-tile GRU-D scan.
//   grid = B/16 workgroups, 256 threads = 8 waves.
//   Wave w owns column tiles {2w, 2w+1} of D=256; keeps h (f32) in C-layout
//   registers across all 48 steps; bf16 h mirror in LDS feeds A-fragments.
//   TDM double-buffers next step's xt/dt tiles into LDS (async, one wave
//   issues, all gate on s_wait_tensorcnt).
// ---------------------------------------------------------------------------
__global__ __launch_bounds__(256) void grud_kernel(
    const float* __restrict__ h0, const int* __restrict__ lens,
    const float* __restrict__ b_ih, const float* __restrict__ b_hh,
    const float* __restrict__ hd_b,
    const __bf16* __restrict__ Xc, const __bf16* __restrict__ Db,
    const __bf16* __restrict__ Wih, const __bf16* __restrict__ Whh,
    const __bf16* __restrict__ hdW,
    float* __restrict__ vslt)
{
    // Manually partitioned LDS (single object => known byte offsets for TDM):
    //   [0   .. 8191 ] hS      : bf16 h-tile mirror [16][256]
    //   [8192.. 10239] xStage  : 2 x bf16 [16][32]
    //   [10240..11263] dStage  : 2 x bf16 [16][16]
    __shared__ __align__(16) __bf16 shmem[16 * D_ + 2 * (16 * 32) + 2 * (16 * 16)];
    __bf16* hS = shmem;
    const unsigned XOFF = 16 * D_ * 2;            // 8192 bytes
    const unsigned DOFF = XOFF + 2 * 16 * 32 * 2; // 10240 bytes

    const int wave = threadIdx.x >> 5;
    const int lane = threadIdx.x & 31;
    const int col  = lane & 15;      // A: M-row / B: N-col / C: N-col
    const int hi   = lane >> 4;      // K-half selector / C row-half
    const int b0   = blockIdx.x * 16;

    const int jb[2] = { (wave * 2) * 16, (wave * 2 + 1) * 16 };

    // per-lane biases for this lane's columns
    float biasR[2], biasZ[2], biasIN[2], biasHN[2], biasDec[2];
#pragma unroll
    for (int j = 0; j < 2; ++j) {
        int n = jb[j] + col;
        biasR[j]   = b_ih[n]        + b_hh[n];
        biasZ[j]   = b_ih[D_ + n]   + b_hh[D_ + n];
        biasIN[j]  = b_ih[2 * D_ + n];
        biasHN[j]  = b_hh[2 * D_ + n];
        biasDec[j] = hd_b[n];
    }

    // constant B-fragments: hd_W (K=16 zero-padded to 32) and W_ih (K=32)
    FragB fhdW[2], fWih[2][3];
    const U4 Z4 = {0u, 0u, 0u, 0u};
#pragma unroll
    for (int j = 0; j < 2; ++j) {
        int n = jb[j] + col;
        fhdW[j].u[0] = *(const U4*)(hdW + n * 16 + hi * 8);
        fhdW[j].u[1] = Z4;
#pragma unroll
        for (int g = 0; g < 3; ++g) {
            int ng = g * D_ + jb[j] + col;
            fWih[j][g].u[0] = *(const U4*)(Wih + ng * 32 + hi * 8);
            fWih[j][g].u[1] = *(const U4*)(Wih + ng * 32 + 16 + hi * 8);
        }
    }

    // sequence lengths for the 8 batch rows this lane's C-registers cover
    int Lr[8];
#pragma unroll
    for (int e = 0; e < 8; ++e) Lr[e] = lens[b0 + e + hi * 8];

    // h in f32 C/D layout: reg e <-> row (e + hi*8), col (jb[j]+col)
    v8f h[2];
#pragma unroll
    for (int j = 0; j < 2; ++j)
#pragma unroll
        for (int e = 0; e < 8; ++e)
            h[j][e] = h0[(size_t)(b0 + e + hi * 8) * D_ + jb[j] + col];

    // TDM prologue: stage t=0 activation tiles into buffer 0
    if (wave == 0) {
        tdm_load_2d_bf16(XOFF, Xc + ((size_t)b0 * T_ + 0) * 32, 32, 16, T_ * 32);
        tdm_load_2d_bf16(DOFF, Db + ((size_t)b0 * T_ + 0) * 16, 16, 16, T_ * 16);
    }

    for (int t = 0; t < T_; ++t) {
        const int buf  = t & 1;
        const __bf16* xS = shmem + 16 * D_ + buf * (16 * 32);
        const __bf16* dS = shmem + 16 * D_ + 2 * (16 * 32) + buf * (16 * 16);

        // ---- async stage t+1, then gate on current step's pair ----
        if (t + 1 < T_) {
            if (wave == 0) {
                const int nb = (t + 1) & 1;
                tdm_load_2d_bf16(XOFF + nb * (16 * 32 * 2),
                                 Xc + ((size_t)b0 * T_ + (t + 1)) * 32, 32, 16, T_ * 32);
                tdm_load_2d_bf16(DOFF + nb * (16 * 16 * 2),
                                 Db + ((size_t)b0 * T_ + (t + 1)) * 16, 16, 16, T_ * 16);
            }
            __builtin_amdgcn_s_wait_tensorcnt(2);   // t's pair done (in-order)
        } else {
            __builtin_amdgcn_s_wait_tensorcnt(0);
        }
        __syncthreads();   // publish staged tiles to all waves

        // ---- decay: h *= exp(-relu(dt @ hd_W^T + hd_b)) ----
        FragB fdt;
        fdt.u[0] = *(const U4*)(dS + col * 16 + hi * 8);
        fdt.u[1] = Z4;
#pragma unroll
        for (int j = 0; j < 2; ++j) {
            v8f a = splat8(biasDec[j]);
            a = wmma_bf16(fdt.v, fhdW[j].v, a);
#pragma unroll
            for (int e = 0; e < 8; ++e)
                h[j][e] *= __expf(-fmaxf(a[e], 0.0f));
        }

        // publish decayed h (bf16) for all waves' A-fragments
#pragma unroll
        for (int j = 0; j < 2; ++j)
#pragma unroll
            for (int e = 0; e < 8; ++e)
                hS[(e + hi * 8) * D_ + jb[j] + col] = (__bf16)h[j][e];
        __syncthreads();

        // ---- gi = xt @ W_ih^T (K=32, one WMMA per tile) ----
        FragB fxt;
        fxt.u[0] = *(const U4*)(xS + col * 32 + hi * 8);
        fxt.u[1] = *(const U4*)(xS + col * 32 + 16 + hi * 8);

        v8f aR[2], aZ[2], aIN[2], aHN[2];
#pragma unroll
        for (int j = 0; j < 2; ++j) {
            aR[j]  = splat8(biasR[j]);
            aZ[j]  = splat8(biasZ[j]);
            aIN[j] = splat8(biasIN[j]);
            aHN[j] = splat8(biasHN[j]);
            aR[j]  = wmma_bf16(fxt.v, fWih[j][0].v, aR[j]);
            aZ[j]  = wmma_bf16(fxt.v, fWih[j][1].v, aZ[j]);
            aIN[j] = wmma_bf16(fxt.v, fWih[j][2].v, aIN[j]);
        }

        // ---- gh = h @ W_hh^T (K=256 -> 8 WMMA K-steps), A shared per kb ----
        // unroll-by-2 (12 WMMAs in flight) keeps VGPR allocation <= 256 so we
        // stay at 4 waves/SIMD; full unroll spilled into the VGPR-MSB range.
#pragma unroll 2
        for (int kb = 0; kb < 8; ++kb) {
            FragB fA;
            const __bf16* hrow = hS + col * D_ + kb * 32 + hi * 8;
            fA.u[0] = *(const U4*)(hrow);
            fA.u[1] = *(const U4*)(hrow + 16);
#pragma unroll
            for (int j = 0; j < 2; ++j) {
                FragB fB;
                const __bf16* wr = Whh + ((size_t)(jb[j] + col)) * D_ + kb * 32 + hi * 8;
                fB.u[0] = *(const U4*)(wr);
                fB.u[1] = *(const U4*)(wr + 16);
                aR[j] = wmma_bf16(fA.v, fB.v, aR[j]);

                wr = Whh + ((size_t)(D_ + jb[j] + col)) * D_ + kb * 32 + hi * 8;
                fB.u[0] = *(const U4*)(wr);
                fB.u[1] = *(const U4*)(wr + 16);
                aZ[j] = wmma_bf16(fA.v, fB.v, aZ[j]);

                wr = Whh + ((size_t)(2 * D_ + jb[j] + col)) * D_ + kb * 32 + hi * 8;
                fB.u[0] = *(const U4*)(wr);
                fB.u[1] = *(const U4*)(wr + 16);
                aHN[j] = wmma_bf16(fA.v, fB.v, aHN[j]);
            }
        }
        __syncthreads();   // protect hS + staged buffers before reuse

        // ---- pointwise GRU update (register-local) + vslt capture ----
#pragma unroll
        for (int j = 0; j < 2; ++j) {
#pragma unroll
            for (int e = 0; e < 8; ++e) {
                float r  = sigmf(aR[j][e]);
                float z  = sigmf(aZ[j][e]);
                float nn = tanhf(aIN[j][e] + r * aHN[j][e]);
                float hn = (1.0f - z) * nn + z * h[j][e];
                h[j][e] = hn;
                if (t == Lr[e] - 1)
                    vslt[(size_t)(b0 + e + hi * 8) * D_ + jb[j] + col] = hn;
            }
        }
    }
}

// ---------------------------------------------------------------------------
// Kernel 4: build padded bf16 classifier input [vslt|img_feat|age|gen|0pad]
// ---------------------------------------------------------------------------
__global__ __launch_bounds__(256) void prep_cls_kernel(
    const float* __restrict__ vslt, const float* __restrict__ img,
    const float* __restrict__ age, const float* __restrict__ gen,
    __bf16* __restrict__ clsb)
{
    size_t i = (size_t)blockIdx.x * 256 + threadIdx.x;
    if (i >= (size_t)B_ * CPAD) return;
    int c = (int)(i % CPAD);
    size_t b = i / CPAD;
    float v;
    if      (c < 256) v = vslt[b * 256 + c];
    else if (c < 512) v = img[b * 256 + (c - 256)];
    else if (c == 512) v = age[b];
    else if (c == 513) v = gen[b];
    else               v = 0.0f;
    clsb[i] = (__bf16)v;
}

// ---------------------------------------------------------------------------
// Kernel 5: K=12-head classifier. grid = K * (B/16), 128 threads = 4 waves.
// ---------------------------------------------------------------------------
__global__ __launch_bounds__(128) void cls_kernel(
    const __bf16* __restrict__ clsb, const __bf16* __restrict__ W1b,
    const float* __restrict__ fc_b1,
    const float* __restrict__ bn_g, const float* __restrict__ bn_b,
    const float* __restrict__ bn_mean, const float* __restrict__ bn_var,
    const float* __restrict__ fc_W2, const float* __restrict__ fc_b2,
    float* __restrict__ out)
{
    __shared__ float red[16];
    const int k  = blockIdx.x >> 8;          // head
    const int b0 = (blockIdx.x & 255) * 16;  // batch tile
    const int wave = threadIdx.x >> 5;
    const int lane = threadIdx.x & 31;
    const int col  = lane & 15;
    const int hi   = lane >> 4;
    const int n    = wave * 16 + col;        // 0..63

    if (threadIdx.x < 16) red[threadIdx.x] = 0.0f;
    __syncthreads();

    v8f acc = splat8(fc_b1[k * H_ + n]);
#pragma unroll
    for (int kb = 0; kb < 17; ++kb) {
        FragB fA, fB;
        const __bf16* ar = clsb + (size_t)(b0 + col) * CPAD + kb * 32 + hi * 8;
        fA.u[0] = *(const U4*)(ar);
        fA.u[1] = *(const U4*)(ar + 16);
        const __bf16* br = W1b + (size_t)(k * H_ + n) * CPAD + kb * 32 + hi * 8;
        fB.u[0] = *(const U4*)(br);
        fB.u[1] = *(const U4*)(br + 16);
        acc = wmma_bf16(fA.v, fB.v, acc);
    }

    const int kn = k * H_ + n;
    float inv = rsqrtf(bn_var[kn] + BN_EPS);
    float mu = bn_mean[kn], g = bn_g[kn], bb = bn_b[kn], w2 = fc_W2[kn];
#pragma unroll
    for (int e = 0; e < 8; ++e) {
        float h1 = fmaxf((acc[e] - mu) * inv * g + bb, 0.0f);
        atomicAdd(&red[e + hi * 8], h1 * w2);   // ds_add_f32 reduce over H
    }
    __syncthreads();
    if (threadIdx.x < 16)
        out[(size_t)k * B_ + b0 + threadIdx.x] =
            sigmf(red[threadIdx.x] + fc_b2[k]);
}

// ---------------------------------------------------------------------------
// Host launcher
// ---------------------------------------------------------------------------
extern "C" void kernel_launch(void* const* d_in, const int* in_sizes, int n_in,
                              void* d_out, int out_size, void* d_ws, size_t ws_size,
                              hipStream_t stream)
{
    const float* x       = (const float*)d_in[0];
    const float* h0      = (const float*)d_in[1];
    const float* m       = (const float*)d_in[2];
    const float* d       = (const float*)d_in[3];
    const float* x_m     = (const float*)d_in[4];
    const float* age     = (const float*)d_in[5];
    const float* gen     = (const float*)d_in[6];
    const float* img     = (const float*)d_in[7];
    const int*   lens    = (const int*)  d_in[8];
    const float* dec_w   = (const float*)d_in[9];
    const float* dec_b   = (const float*)d_in[10];
    const float* hd_W    = (const float*)d_in[11];
    const float* hd_b    = (const float*)d_in[12];
    const float* W_ih    = (const float*)d_in[13];
    const float* W_hh    = (const float*)d_in[14];
    const float* b_ih    = (const float*)d_in[15];
    const float* b_hh    = (const float*)d_in[16];
    const float* fc_W1   = (const float*)d_in[17];
    const float* fc_b1   = (const float*)d_in[18];
    const float* bn_g    = (const float*)d_in[19];
    const float* bn_b    = (const float*)d_in[20];
    const float* bn_mean = (const float*)d_in[21];
    const float* bn_var  = (const float*)d_in[22];
    const float* fc_W2   = (const float*)d_in[23];
    const float* fc_b2   = (const float*)d_in[24];
    float* out = (float*)d_out;
    (void)in_sizes; (void)n_in; (void)out_size; (void)ws_size;

    // workspace layout (bytes, all 256-aligned)
    char* w = (char*)d_ws;
    const size_t szXc   = (size_t)B_ * T_ * 32 * 2;   // 12,582,912
    const size_t szDb   = (size_t)B_ * T_ * 16 * 2;   //  6,291,456
    const size_t szWih  = (size_t)G_ * 32 * 2;        //     49,152
    const size_t szWhh  = (size_t)G_ * D_ * 2;        //    393,216
    const size_t szhdW  = (size_t)D_ * F_ * 2;        //      8,192
    const size_t szW1b  = (size_t)K_ * H_ * CPAD * 2; //    835,584
    const size_t szVslt = (size_t)B_ * D_ * 4;        //  4,194,304
    __bf16* Xc   = (__bf16*)(w);
    __bf16* Db   = (__bf16*)(w + szXc);
    __bf16* Wih  = (__bf16*)(w + szXc + szDb);
    __bf16* Whh  = (__bf16*)(w + szXc + szDb + szWih);
    __bf16* hdW  = (__bf16*)(w + szXc + szDb + szWih + szWhh);
    __bf16* W1b  = (__bf16*)(w + szXc + szDb + szWih + szWhh + szhdW);
    float*  vslt = (float*) (w + szXc + szDb + szWih + szWhh + szhdW + szW1b);
    __bf16* clsb = (__bf16*)(w + szXc + szDb + szWih + szWhh + szhdW + szW1b + szVslt);

    {   // 1. elementwise preprocessing -> bf16 activations
        size_t n = (size_t)B_ * T_ * F_;
        prep_x_kernel<<<dim3((unsigned)((n + 255) / 256)), dim3(256), 0, stream>>>(
            x, m, d, x_m, dec_w, dec_b, Xc, Db);
    }
    {   // 2. weight conversion -> bf16
        int n = G_ * 32 + G_ * D_ + D_ * F_ + K_ * H_ * CPAD;
        prep_w_kernel<<<dim3((n + 255) / 256), dim3(256), 0, stream>>>(
            W_ih, W_hh, hd_W, fc_W1, Wih, Whh, hdW, W1b);
    }
    {   // 3. the recurrent scan (256 batch tiles x 48 steps, WMMA + TDM)
        grud_kernel<<<dim3(B_ / 16), dim3(256), 0, stream>>>(
            h0, lens, b_ih, b_hh, hd_b, Xc, Db, Wih, Whh, hdW, vslt);
    }
    {   // 4. classifier input assembly (bf16, zero-padded to 544)
        size_t n = (size_t)B_ * CPAD;
        prep_cls_kernel<<<dim3((unsigned)((n + 255) / 256)), dim3(256), 0, stream>>>(
            vslt, img, age, gen, clsb);
    }
    {   // 5. 12-head classifier
        cls_kernel<<<dim3(K_ * (B_ / 16)), dim3(128), 0, stream>>>(
            clsb, W1b, fc_b1, bn_g, bn_b, bn_mean, bn_var, fc_W2, fc_b2, out);
    }
}
